// GumbelGraphNetworkClf_48799418417234
// MI455X (gfx1250) — compile-verified, matrix operand order checked by reading, not run on hardware
//
#include <hip/hip_runtime.h>
#include <hip/hip_bf16.h>

typedef __attribute__((ext_vector_type(16))) _Float16 v16h;
typedef __attribute__((ext_vector_type(8)))  float    v8f;

#define Bq 4
#define Nn 512
#define Dd 4
#define Hh 256

// workspace byte offsets
static constexpr size_t AROW_OFF = 0;                               // 2 MB  (B*N x H f32)
static constexpr size_t CROW_OFF = 2u * 1024 * 1024;                // 2 MB
static constexpr size_t WB_OFF   = 4u * 1024 * 1024;                // 128 KB (W_e2 f16 swizzled)
static constexpr size_t AGG_OFF  = 4u * 1024 * 1024 + 256u * 1024;  // 2 MB  (agg f32)

// ---------------------------------------------------------------------------
// A[b,j,h] = b_e1[h] + sum_d x[b,j,d] * W_e1[d,h]        (innode half)
// C[b,i,h] =           sum_d x[b,i,d] * W_e1[D+d,h]      (outnode half)
// ---------------------------------------------------------------------------
__global__ void __launch_bounds__(256)
prep_rows(const float* __restrict__ x, const float* __restrict__ We1,
          const float* __restrict__ be1, float* __restrict__ Arow,
          float* __restrict__ Crow) {
  int row = blockIdx.x;        // b*N + node
  int h   = threadIdx.x;       // 0..255
  float xv[Dd];
#pragma unroll
  for (int d = 0; d < Dd; ++d) xv[d] = x[row * Dd + d];
  float a = be1[h], c = 0.f;
#pragma unroll
  for (int d = 0; d < Dd; ++d) {
    a += xv[d] * We1[d * Hh + h];
    c += xv[d] * We1[(Dd + d) * Hh + h];
  }
  Arow[row * Hh + h] = a;
  Crow[row * Hh + h] = c;
}

// ---------------------------------------------------------------------------
// Swizzle W_e2 (f32 [256,256]) into f16 WMMA B-fragment layout.
// Fragment (t = h'-tile 0..15, c = k-chunk 0..7, lane 0..31) = 16 contiguous
// halfs. Lane l<16: column N=l, K order {0..7,16..23}; lane l>=16: N=l-16,
// K order {8..15,24..31}  (mirrors the ISA 16-bit A-matrix 16x32 layout).
// ---------------------------------------------------------------------------
__global__ void __launch_bounds__(256)
prep_wb(const float* __restrict__ We2, _Float16* __restrict__ Wb) {
  int idx  = blockIdx.x * 256 + threadIdx.x;   // 65536 entries
  int p    = idx & 15;
  int lane = (idx >> 4) & 31;
  int c    = (idx >> 9) & 7;
  int t    = idx >> 12;
  int kl;
  if (lane < 16) kl = (p < 8) ? p : p + 8;
  else           kl = (p < 8) ? p + 8 : p + 16;
  int k = 32 * c + kl;
  int n = 16 * t + (lane & 15);
  Wb[idx] = (_Float16)We2[k * Hh + n];
}

// ---------------------------------------------------------------------------
// Fused edge pipeline: one workgroup per (b,j).
// agg[b,j,h'] = sum_i adj[b,i,j] * relu( relu(A[b,j]+C[b,i]) @ W_e2 + b_e2 )
// ---------------------------------------------------------------------------
__global__ void __launch_bounds__(256)
edge_fused(const float* __restrict__ adj, const float* __restrict__ Arow,
           const float* __restrict__ Crow, const _Float16* __restrict__ Wb,
           const float* __restrict__ be2, float* __restrict__ agg) {
  // 64 i-rows x 256 k, stored as WMMA A-fragments: [m-tile][k-chunk][lane][16]
  __shared__ __align__(32) _Float16 sN2E[4][8][32][16];   // 32 KB
  __shared__ float sAdj[64];

  const int bj   = blockIdx.x;
  const int b    = bj >> 9;
  const int j    = bj & (Nn - 1);
  const int tid  = threadIdx.x;
  const int lane = tid & 31;
  const int w    = tid >> 5;           // wave 0..7
  const int hi   = lane >> 4;          // half-wave (rows 0-7 vs 8-15 of D tile)
  const int col  = lane & 15;
  const int t0   = 2 * w, t1 = 2 * w + 1;   // this wave's two h'-tiles

  // hoist B fragments (invariant over i-chunks and m-tiles): 128 VGPRs
  const v16h* wb = (const v16h*)Wb;
  v16h b0f[8], b1f[8];
#pragma unroll
  for (int c = 0; c < 8; ++c) {
    b0f[c] = wb[(t0 * 8 + c) * 32 + lane];
    b1f[c] = wb[(t1 * 8 + c) * 32 + lane];
  }
  const float bias0 = be2[16 * t0 + col];
  const float bias1 = be2[16 * t1 + col];

  // per-thread fill constants: this thread owns feature h = tid for the fill
  const int   hc    = tid;
  const int   kc    = hc >> 5;
  const int   kk    = hc & 31;
  const int   fl_hi = (kk >> 3) & 1;
  const int   p     = (kk & 7) | (((kk >> 4) & 1) << 3);
  const float aj    = Arow[bj * Hh + hc];
  const float* crow_base = Crow + (size_t)b * Nn * Hh + hc;

  float acc0 = 0.f, acc1 = 0.f;

  for (int ic = 0; ic < 8; ++ic) {
    const int i0 = ic * 64;
    __syncthreads();   // previous chunk fully consumed before refill
    // build relu(A_j + C_i) chunk in A-fragment layout (f16)
#pragma unroll 4
    for (int r = 0; r < 64; ++r) {
      float v = aj + crow_base[(size_t)(i0 + r) * Hh];
      v = v > 0.f ? v : 0.f;
      sN2E[r >> 4][kc][(r & 15) + (fl_hi << 4)][p] = (_Float16)v;
    }
    if (tid < 64) sAdj[tid] = adj[((size_t)b * Nn + i0 + tid) * Nn + j];
    __syncthreads();

    for (int mt = 0; mt < 4; ++mt) {
      v8f d0 = {}, d1 = {};
#pragma unroll
      for (int c = 0; c < 8; ++c) {
        v16h a = *(const v16h*)&sN2E[mt][c][lane][0];
        d0 = __builtin_amdgcn_wmma_f32_16x16x32_f16(false, a, false, b0f[c],
                                                    (short)0, d0, false, false);
        d1 = __builtin_amdgcn_wmma_f32_16x16x32_f16(false, a, false, b1f[c],
                                                    (short)0, d1, false, false);
      }
      // bias + relu + adj-weighted accumulation over the 16 rows of this tile
#pragma unroll
      for (int v = 0; v < 8; ++v) {
        float av = sAdj[mt * 16 + v + 8 * hi];
        float e0 = d0[v] + bias0; e0 = e0 > 0.f ? e0 : 0.f;
        float e1 = d1[v] + bias1; e1 = e1 > 0.f ? e1 : 0.f;
        acc0 += av * e0;
        acc1 += av * e1;
      }
    }
  }

  // combine the two half-wave row groups (SWAPX16 ds_swizzle)
  float o0 = __int_as_float(__builtin_amdgcn_ds_swizzle(__float_as_int(acc0), 0x401F));
  float o1 = __int_as_float(__builtin_amdgcn_ds_swizzle(__float_as_int(acc1), 0x401F));
  acc0 += o0;
  acc1 += o1;
  if (lane < 16) {
    agg[(size_t)bj * Hh + 16 * t0 + lane] = acc0;
    agg[(size_t)bj * Hh + 16 * t1 + lane] = acc1;
  }
}

// ---------------------------------------------------------------------------
// Per-node tail: out1=relu(agg@Wn1+b); out2=relu(out1@Wn2+b);
// out4=[x,out2]@Wo1+b; out5=out4@Wo+b; log_softmax over D=4.
// One workgroup (256 threads) per node row.
// ---------------------------------------------------------------------------
__global__ void __launch_bounds__(256)
node_mlp(const float* __restrict__ x, const float* __restrict__ agg,
         const float* __restrict__ Wn1, const float* __restrict__ bn1,
         const float* __restrict__ Wn2, const float* __restrict__ bn2,
         const float* __restrict__ Wo1, const float* __restrict__ bo1,
         const float* __restrict__ Wo,  const float* __restrict__ bo,
         float* __restrict__ out) {
  __shared__ float sA[Hh];
  __shared__ float sB[Hh];
  __shared__ float s5[4];
  __shared__ float sred[2];
  const int r = blockIdx.x;      // b*N + j
  const int t = threadIdx.x;

  sA[t] = agg[(size_t)r * Hh + t];
  __syncthreads();

  float acc = bn1[t];
  for (int k = 0; k < Hh; ++k) acc += sA[k] * Wn1[k * Hh + t];
  acc = acc > 0.f ? acc : 0.f;
  sB[t] = acc;
  __syncthreads();

  acc = bn2[t];
  for (int k = 0; k < Hh; ++k) acc += sB[k] * Wn2[k * Hh + t];
  acc = acc > 0.f ? acc : 0.f;
  sA[t] = acc;
  __syncthreads();

  acc = bo1[t];
#pragma unroll
  for (int d = 0; d < Dd; ++d) acc += x[r * Dd + d] * Wo1[d * Hh + t];
  for (int k = 0; k < Hh; ++k) acc += sA[k] * Wo1[(Dd + k) * Hh + t];
  sB[t] = acc;                   // out4 (no relu)
  __syncthreads();

  if (t < 4) {
    float o = bo[t];
    for (int k = 0; k < Hh; ++k) o += sB[k] * Wo[k * Dd + t];
    s5[t] = o;
  }
  __syncthreads();
  if (t == 0) {
    float m = s5[0];
    for (int d = 1; d < 4; ++d) m = fmaxf(m, s5[d]);
    float sum = 0.f;
    for (int d = 0; d < 4; ++d) sum += __expf(s5[d] - m);
    sred[0] = m;
    sred[1] = __logf(sum);
  }
  __syncthreads();
  if (t < 4) out[(size_t)r * Dd + t] = s5[t] - sred[0] - sred[1];
}

// ---------------------------------------------------------------------------
extern "C" void kernel_launch(void* const* d_in, const int* in_sizes, int n_in,
                              void* d_out, int out_size, void* d_ws, size_t ws_size,
                              hipStream_t stream) {
  const float* x   = (const float*)d_in[0];
  const float* adj = (const float*)d_in[1];
  const float* We1 = (const float*)d_in[2];
  const float* be1 = (const float*)d_in[3];
  const float* We2 = (const float*)d_in[4];
  const float* be2 = (const float*)d_in[5];
  const float* Wn1 = (const float*)d_in[6];
  const float* bn1 = (const float*)d_in[7];
  const float* Wn2 = (const float*)d_in[8];
  const float* bn2 = (const float*)d_in[9];
  const float* Wo1 = (const float*)d_in[10];
  const float* bo1 = (const float*)d_in[11];
  const float* Wo  = (const float*)d_in[12];
  const float* bo  = (const float*)d_in[13];

  char* ws = (char*)d_ws;
  float*    Arow = (float*)(ws + AROW_OFF);
  float*    Crow = (float*)(ws + CROW_OFF);
  _Float16* Wb   = (_Float16*)(ws + WB_OFF);
  float*    aggp = (float*)(ws + AGG_OFF);

  prep_rows<<<Bq * Nn, 256, 0, stream>>>(x, We1, be1, Arow, Crow);
  prep_wb<<<256, 256, 0, stream>>>(We2, Wb);
  edge_fused<<<Bq * Nn, 256, 0, stream>>>(adj, Arow, Crow, Wb, be2, aggp);
  node_mlp<<<Bq * Nn, 256, 0, stream>>>(x, aggp, Wn1, bn1, Wn2, bn2,
                                        Wo1, bo1, Wo, bo, (float*)d_out);
}
